// ValueNetworkLSTM_22016002359302
// MI455X (gfx1250) — matrix operand order repaired and checked by reading, not compiled, not run on hardware
//
#include <hip/hip_runtime.h>
#include <hip/hip_bf16.h>

typedef __bf16 bf16_t;
typedef bf16_t v16bf __attribute__((ext_vector_type(16)));
typedef float  v8f   __attribute__((ext_vector_type(8)));

#define DEVFN static __device__ __forceinline__

constexpr int kT  = 128;
constexpr int kB  = 256;
constexpr int kF  = 512;
constexpr int kH  = 1024;
constexpr int kG  = 4 * kH;   // 4096
constexpr int kTB = kT * kB;  // 32768

// padded LDS row stride (elements) for a 32-wide k-chunk: keeps 16B alignment,
// spreads rows across banks (80B step -> 16 distinct 4-bank groups)
constexpr int kLds = 40;

DEVFN float lrelu(float x) { return x >= 0.f ? x : 0.01f * x; }
DEVFN float sigmoidf(float x) { return 1.f / (1.f + __expf(-x)); }

union FragBf { uint4 u[2]; v16bf v; };

// ---------------- gfx1250 async global->LDS copy (ASYNCcnt path) ------------
#if defined(__gfx1250__) && __has_builtin(__builtin_amdgcn_global_load_async_to_lds_b128)
#define HAVE_ASYNC_LDS 1
#else
#define HAVE_ASYNC_LDS 0
#endif

typedef int v4i_vs __attribute__((vector_size(16)));
typedef __attribute__((address_space(1))) v4i_vs* g_v4p;
typedef __attribute__((address_space(3))) v4i_vs* l_v4p;

DEVFN void cp16_g2l(void* l, const void* g) {
#if HAVE_ASYNC_LDS
  __builtin_amdgcn_global_load_async_to_lds_b128((g_v4p)(g), (l_v4p)(l), 0, 0);
#else
  *(uint4*)l = *(const uint4*)g;
#endif
}

DEVFN void async_wait() {
#if HAVE_ASYNC_LDS
#if __has_builtin(__builtin_amdgcn_s_wait_asynccnt)
  __builtin_amdgcn_s_wait_asynccnt(0);
#else
  asm volatile("s_wait_asynccnt 0" ::: "memory");
#endif
#endif
}

// ------------------------- conversion / init kernels ------------------------
__global__ void cvt_kernel(const float* __restrict__ s, bf16_t* __restrict__ d, long n) {
  long i = (long)blockIdx.x * blockDim.x + threadIdx.x;
  if (i < n) d[i] = (bf16_t)s[i];
}

__global__ void init_state_kernel(bf16_t* __restrict__ h, float* __restrict__ c, int n) {
  int i = blockIdx.x * blockDim.x + threadIdx.x;
  if (i < n) { h[i] = (bf16_t)0.f; c[i] = 0.f; }
}

// ------------------------- big WMMA GEMM ------------------------------------
// out[M,N] = epilogue(A[M,K] @ W[N,K]^T + bias); A row-major bf16, W row-major bf16.
// Block: 256 threads = 8 waves; block tile = 16(M) x 128(N); wave tile = 16x16.
template <int K, int N, bool ACT, bool TWO_BIAS, bool OUT_BF16>
__global__ __launch_bounds__(256) void wmma_gemm_kernel(
    const bf16_t* __restrict__ A, const bf16_t* __restrict__ W,
    const float* __restrict__ bias0, const float* __restrict__ bias1,
    float* __restrict__ outF, bf16_t* __restrict__ outB) {
  __shared__ alignas(32) bf16_t lsA[16 * kLds];    // A tile 16 x 32 (row-major, padded)
  __shared__ alignas(32) bf16_t lsB[128 * kLds];   // W rows: 128 x 32 k-chunk

  const int tid  = threadIdx.x;
  const int w    = tid >> 5;
  const int lane = tid & 31;
  const int half = lane >> 4;
  const int col  = lane & 15;
  const long mbase = (long)blockIdx.y * 16;
  const long nbase = (long)blockIdx.x * 128;

  v8f acc = (v8f)(0.0f);

  for (int kb = 0; kb < K; kb += 32) {
    if (tid < 64) {  // stage A tile: 16 rows x 32 cols, one uint4 per thread
      int m = tid >> 2, seg = tid & 3;
      *(uint4*)&lsA[m * kLds + seg * 8] =
          *(const uint4*)&A[(mbase + m) * K + kb + seg * 8];
    }
    // stage W rows nbase..nbase+127, k-chunk [kb, kb+32): async DMA to LDS
    for (int i = tid; i < 512; i += 256) {
      int nl = i >> 2, seg = i & 3;
      cp16_g2l(&lsB[nl * kLds + seg * 8],
               &W[(long)(nbase + nl) * K + kb + seg * 8]);
    }
    async_wait();
    __syncthreads();

    FragBf fa, fb;
    fa.u[0] = *(const uint4*)&lsA[col * kLds + half * 8];
    fa.u[1] = *(const uint4*)&lsA[col * kLds + 16 + half * 8];
    fb.u[0] = *(const uint4*)&lsB[(w * 16 + col) * kLds + half * 8];
    fb.u[1] = *(const uint4*)&lsB[(w * 16 + col) * kLds + 16 + half * 8];
    acc = __builtin_amdgcn_wmma_f32_16x16x32_bf16(false, fa.v, false, fb.v,
                                                  (short)0, acc, false, false);
    __syncthreads();
  }

  const long n = nbase + w * 16 + col;
  float bias = bias0[n] + (TWO_BIAS ? bias1[n] : 0.f);
#pragma unroll
  for (int r = 0; r < 8; ++r) {
    long m = mbase + half * 8 + r;
    float v = acc[r] + bias;
    if (ACT) v = lrelu(v);
    if (OUT_BF16) outB[m * N + n] = (bf16_t)v;
    else          outF[m * N + n] = v;
  }
}

// ------------------------- fused LSTM step ----------------------------------
// gates[B,4H] = xg[t] + (h*mask) @ W_hh^T ; then elementwise LSTM cell update.
// Each wave holds 4 accumulators (i,f,g,o) for a 16(batch) x 16(hidden) tile.
__global__ __launch_bounds__(256) void lstm_step_kernel(
    int t,
    const bf16_t* __restrict__ hin,   // [B,H] bf16 (prev h)
    bf16_t*       __restrict__ hout,  // [B,H] bf16 (new h)
    float*        __restrict__ cbuf,  // [B,H] f32 (cell state, in/out)
    const bf16_t* __restrict__ Whh,   // [4H,H] bf16 (natural torch layout)
    const float*  __restrict__ xg,    // [T,B,4H] f32 (x gates incl. biases)
    const int*    __restrict__ dones, // [T,B]
    float*        __restrict__ hs)    // [T,B,H] f32 (h outputs)
{
  __shared__ alignas(32) bf16_t lsA[16 * kLds];    // masked h tile 16 x 32
  __shared__ alignas(32) bf16_t lsB[512 * kLds];   // 4 gates x 128 W_hh rows x 32 k

  const int tid  = threadIdx.x;
  const int w    = tid >> 5;
  const int lane = tid & 31;
  const int half = lane >> 4;
  const int col  = lane & 15;
  const int mbase  = blockIdx.y * 16;    // batch tile
  const int nblock = blockIdx.x * 128;   // hidden tile (within H)

  v8f acc[4];
#pragma unroll
  for (int q = 0; q < 4; ++q) acc[q] = (v8f)(0.0f);

  for (int kb = 0; kb < kH; kb += 32) {
    if (tid < 64) {  // stage masked h tile
      int m = tid >> 2, seg = tid & 3;
      int done = dones[t * kB + mbase + m];
      uint4 v = *(const uint4*)&hin[(long)(mbase + m) * kH + kb + seg * 8];
      uint4 z = make_uint4(0u, 0u, 0u, 0u);
      *(uint4*)&lsA[m * kLds + seg * 8] = done ? z : v;
    }
    // stage W_hh rows for 4 gate blocks: rows q*H + nblock + (0..127), k-chunk 32
    for (int i = tid; i < 2048; i += 256) {
      int nl = i >> 2;            // 0..511 : q*128 + r
      int seg = i & 3;
      int q = nl >> 7, r = nl & 127;
      cp16_g2l(&lsB[nl * kLds + seg * 8],
               &Whh[((long)q * kH + nblock + r) * kH + kb + seg * 8]);
    }
    async_wait();
    __syncthreads();

    FragBf fa;
    fa.u[0] = *(const uint4*)&lsA[col * kLds + half * 8];
    fa.u[1] = *(const uint4*)&lsA[col * kLds + 16 + half * 8];
#pragma unroll
    for (int q = 0; q < 4; ++q) {
      FragBf fb;
      int row = q * 128 + w * 16 + col;
      fb.u[0] = *(const uint4*)&lsB[row * kLds + half * 8];
      fb.u[1] = *(const uint4*)&lsB[row * kLds + 16 + half * 8];
      acc[q] = __builtin_amdgcn_wmma_f32_16x16x32_bf16(false, fa.v, false, fb.v,
                                                       (short)0, acc[q], false, false);
    }
    __syncthreads();
  }

  const int n = nblock + w * 16 + col;  // hidden index
#pragma unroll
  for (int r = 0; r < 8; ++r) {
    int m = mbase + half * 8 + r;
    long rowg = ((long)t * kB + m) * (long)kG;
    float gi = sigmoidf(acc[0][r] + xg[rowg + 0 * kH + n]);
    float gf = sigmoidf(acc[1][r] + xg[rowg + 1 * kH + n]);
    float gg = tanhf   (acc[2][r] + xg[rowg + 2 * kH + n]);
    float go = sigmoidf(acc[3][r] + xg[rowg + 3 * kH + n]);
    int done = dones[t * kB + m];
    float c0v = done ? 0.f : cbuf[(long)m * kH + n];
    float cn  = gf * c0v + gi * gg;
    float hn  = go * tanhf(cn);
    cbuf[(long)m * kH + n] = cn;
    hout[(long)m * kH + n] = (bf16_t)hn;
    hs[((long)t * kB + m) * (long)kH + n] = hn;
  }
}

// ------------------------- value head ---------------------------------------
// value[t,b] = leakyrelu(hs[t,b,:]) . Wv + bv ; one wave per row.
__global__ __launch_bounds__(256) void value_kernel(
    const float* __restrict__ hs, const float* __restrict__ Wv,
    const float* __restrict__ bv, float* __restrict__ out) {
  int row  = blockIdx.x * 8 + (threadIdx.x >> 5);
  int lane = threadIdx.x & 31;
  const float* hrow = hs + (long)row * kH;
  float s = 0.f;
  for (int j = lane * 4; j < kH; j += 128) {
    float4 hv = *(const float4*)&hrow[j];
    float4 wv = *(const float4*)&Wv[j];
    s += lrelu(hv.x) * wv.x + lrelu(hv.y) * wv.y +
         lrelu(hv.z) * wv.z + lrelu(hv.w) * wv.w;
  }
#pragma unroll
  for (int off = 16; off > 0; off >>= 1) s += __shfl_down(s, off, 32);
  if (lane == 0) out[row] = s + bv[0];
}

// hidden = stack(hT, cT); hT == hs[T-1]
__global__ void tail_kernel(const float* __restrict__ hs, const float* __restrict__ c,
                            float* __restrict__ out) {
  int i = blockIdx.x * blockDim.x + threadIdx.x;
  const int nBH = kB * kH;
  if (i < nBH) {
    out[kTB + i]       = hs[(long)(kT - 1) * nBH + i];
    out[kTB + nBH + i] = c[i];
  }
}

// ------------------------- launcher -----------------------------------------
extern "C" void kernel_launch(void* const* d_in, const int* in_sizes, int n_in,
                              void* d_out, int out_size, void* d_ws, size_t ws_size,
                              hipStream_t stream) {
  const float* x     = (const float*)d_in[0];
  const int*   dones = (const int*)  d_in[1];
  const float* W1    = (const float*)d_in[2];
  const float* b1    = (const float*)d_in[3];
  const float* W_ih  = (const float*)d_in[4];
  const float* W_hh  = (const float*)d_in[5];
  const float* b_ih  = (const float*)d_in[6];
  const float* b_hh  = (const float*)d_in[7];
  const float* Wv    = (const float*)d_in[8];
  const float* bv    = (const float*)d_in[9];
  float* out = (float*)d_out;

  char* ws = (char*)d_ws;
  auto alloc = [&](size_t bytes) {
    char* p = ws;
    ws += (bytes + 255) & ~(size_t)255;
    return p;
  };
  bf16_t* Xbf   = (bf16_t*)alloc((size_t)kTB * kF * 2);
  bf16_t* W1b   = (bf16_t*)alloc((size_t)kH * kF * 2);   // [H,F] natural
  bf16_t* Wihb  = (bf16_t*)alloc((size_t)kG * kH * 2);   // [4H,H] natural
  bf16_t* Whhb  = (bf16_t*)alloc((size_t)kG * kH * 2);   // [4H,H] natural
  bf16_t* Sbf   = (bf16_t*)alloc((size_t)kTB * kH * 2);
  float*  XG    = (float*) alloc((size_t)kTB * kG * 4);
  bf16_t* hbuf0 = (bf16_t*)alloc((size_t)kB * kH * 2);
  bf16_t* hbuf1 = (bf16_t*)alloc((size_t)kB * kH * 2);
  float*  cbuf  = (float*) alloc((size_t)kB * kH * 4);
  float*  hs    = (float*) alloc((size_t)kTB * kH * 4);

  // 1) fp32 -> bf16 conversions (one-shot, bandwidth-bound; weights keep layout)
  cvt_kernel<<<(int)(((long)kTB * kF + 255) / 256), 256, 0, stream>>>(x, Xbf, (long)kTB * kF);
  cvt_kernel<<<(int)(((long)kH * kF + 255) / 256), 256, 0, stream>>>(W1, W1b, (long)kH * kF);
  cvt_kernel<<<(int)(((long)kG * kH + 255) / 256), 256, 0, stream>>>(W_ih, Wihb, (long)kG * kH);
  cvt_kernel<<<(int)(((long)kG * kH + 255) / 256), 256, 0, stream>>>(W_hh, Whhb, (long)kG * kH);

  // 2) shared = leakyrelu(x @ W1^T + b1)  -> bf16
  wmma_gemm_kernel<kF, kH, true, false, true>
      <<<dim3(kH / 128, kTB / 16), 256, 0, stream>>>(Xbf, W1b, b1, nullptr, nullptr, Sbf);

  // 3) x_gates = shared @ W_ih^T + b_ih + b_hh  -> f32
  wmma_gemm_kernel<kH, kG, false, true, false>
      <<<dim3(kG / 128, kTB / 16), 256, 0, stream>>>(Sbf, Wihb, b_ih, b_hh, XG, nullptr);

  // 4) recurrent scan (ping-pong h buffers)
  init_state_kernel<<<(kB * kH + 255) / 256, 256, 0, stream>>>(hbuf0, cbuf, kB * kH);
  for (int t = 0; t < kT; ++t) {
    bf16_t* hin  = (t & 1) ? hbuf1 : hbuf0;
    bf16_t* hout = (t & 1) ? hbuf0 : hbuf1;
    lstm_step_kernel<<<dim3(kH / 128, kB / 16), 256, 0, stream>>>(
        t, hin, hout, cbuf, Whhb, XG, dones, hs);
  }

  // 5) value head + hidden-state outputs
  value_kernel<<<kTB / 8, 256, 0, stream>>>(hs, Wv, bv, out);
  tail_kernel<<<(kB * kH + 255) / 256, 256, 0, stream>>>(hs, cbuf, out);
}
